// CognitiveProcessor_55886114455824
// MI455X (gfx1250) — matrix operand order repaired and checked by reading, not compiled, not run on hardware
//
#include <hip/hip_runtime.h>

// ---------------------------------------------------------------------------
// MI455X (gfx1250) fused "CognitiveProcessor" kernel.
// bf16 WMMA (v_wmma_f32_16x16x32_bf16) for all 4 GEMMs, f32 accumulate.
// Memory-bound by design: 384 MB irreducible HBM traffic @ 23.3 TB/s.
// Round 3: branch-free fast tanh (v_exp_f32 + v_rcp_f32) replaces OCML
// tanhf, which was emitting divergent saveexec/cbranch sequences between
// GEMM1 and GEMM2.
// ---------------------------------------------------------------------------

#define E_DIM   512
#define O_DIM   64
#define DT_STEP 0.01f
#define N_STEPS 10
#define TM      128     // rows per workgroup (8 waves x 16 rows)
#define T_STRIDE 72     // padded bf16 row stride for t/phases tile (bank spread)
#define H_STRIDE 520    // padded bf16 row stride for h tile (bank spread)

typedef __attribute__((ext_vector_type(8)))  float        v8f;
typedef __attribute__((ext_vector_type(8)))  unsigned int v8u;
typedef __attribute__((ext_vector_type(16))) __bf16       v16bf;
typedef __attribute__((ext_vector_type(2)))  __bf16       v2bf;

// ---- f32 pair -> packed bf16 dword via hardware cvt ----------------------
__device__ __forceinline__ unsigned int pk_bf16(float lo, float hi) {
    v2bf v;
    v.x = (__bf16)lo;   // v_cvt_pk_bf16_f32
    v.y = (__bf16)hi;
    return __builtin_bit_cast(unsigned int, v);
}

// ---- branch-free tanh: (e^{2x}-1)/(e^{2x}+1), clamped --------------------
// Straight-line: 2x clamp, v_exp_f32, v_rcp_f32, mul/add. No EXEC churn.
__device__ __forceinline__ float fast_tanh(float x) {
    x = fminf(fmaxf(x, -10.0f), 10.0f);
    float t = __expf(2.0f * x);
    return (t - 1.0f) * __builtin_amdgcn_rcpf(t + 1.0f);
}

// ---- WMMA wrapper -------------------------------------------------------
__device__ __forceinline__ v8f wmma_bf16(v8u A, v8u B, v8f C) {
    return __builtin_amdgcn_wmma_f32_16x16x32_bf16(
        false, __builtin_bit_cast(v16bf, A),
        false, __builtin_bit_cast(v16bf, B),
        (short)0, C, false, false);
}

// ---- A fragment (16x32 bf16) from a global f32 row ----------------------
// ISA layout: lanes 0-15 hold K {0..7,16..23}, lanes 16-31 hold K {8..15,24..31}
// khalf = 0 (lanes 0-15) or 8 (lanes 16-31); rowp points at this k-chunk.
__device__ __forceinline__ v8u a_frag_global(const float* rowp, int khalf) {
    const float4* p0 = (const float4*)(rowp + khalf);
    const float4* p1 = (const float4*)(rowp + 16 + khalf);
    float4 f0 = p0[0], f1 = p0[1];
    float4 f2 = p1[0], f3 = p1[1];
    v8u r;
    r[0] = pk_bf16(f0.x, f0.y); r[1] = pk_bf16(f0.z, f0.w);
    r[2] = pk_bf16(f1.x, f1.y); r[3] = pk_bf16(f1.z, f1.w);
    r[4] = pk_bf16(f2.x, f2.y); r[5] = pk_bf16(f2.z, f2.w);
    r[6] = pk_bf16(f3.x, f3.y); r[7] = pk_bf16(f3.z, f3.w);
    return r;
}

// ---- A fragment from an LDS bf16 row (row-major) ------------------------
__device__ __forceinline__ v8u a_frag_lds(const __bf16* rowp, int khalf) {
    uint4 a = *(const uint4*)(rowp + khalf);        // K khalf..khalf+7
    uint4 b = *(const uint4*)(rowp + 16 + khalf);   // K 16+khalf..16+khalf+7
    v8u r;
    r[0] = a.x; r[1] = a.y; r[2] = a.z; r[3] = a.w;
    r[4] = b.x; r[5] = b.y; r[6] = b.z; r[7] = b.w;
    return r;
}

// ---- B fragment (32x16 bf16) from n-major (transposed) weights ----------
// ISA layout: lanes 0-15 hold K {0..7,16..23}, lanes 16-31 hold K {8..15,24..31}.
// Wt[n][k]; kbase already includes the per-half +16.
__device__ __forceinline__ v8u b_frag(const __bf16* Wt, int n, int Kdim,
                                      int kbase) {
    const __bf16* q = Wt + (size_t)n * Kdim + (kbase & ~31) + ((kbase & 16) >> 1);
    uint4 a = *(const uint4*)q, b = *(const uint4*)(q + 16);
    v8u r;
    r[0] = a.x; r[1] = a.y; r[2] = a.z; r[3] = a.w;
    r[4] = b.x; r[5] = b.y; r[6] = b.z; r[7] = b.w;
    return r;
}

// ---------------------------------------------------------------------------
// Weight prep: f32 -> bf16, transposed to n-major so B-fragments are two
// contiguous 16-byte loads per lane. Lives in d_ws (L2-resident, 648 KB).
//   W1t [64][512]  W2t [64][64]  W3t [512][64]  W4t [512][512]
// ---------------------------------------------------------------------------
#define W1T_OFF 0
#define W2T_OFF (64 * 512)
#define W3T_OFF (W2T_OFF + 64 * 64)
#define W4T_OFF (W3T_OFF + 512 * 64)

__global__ void __launch_bounds__(256)
convert_weights(const float* __restrict__ W1, const float* __restrict__ W2,
                const float* __restrict__ W3, const float* __restrict__ W4,
                __bf16* __restrict__ ws) {
    int i = blockIdx.x * 256 + threadIdx.x;
    if (i < 64 * 512)  { int n = i / 512, k = i % 512; ws[W1T_OFF + i] = (__bf16)W1[k * 64  + n]; }
    if (i < 64 * 64)   { int n = i / 64,  k = i % 64;  ws[W2T_OFF + i] = (__bf16)W2[k * 64  + n]; }
    if (i < 512 * 64)  { int n = i / 64,  k = i % 64;  ws[W3T_OFF + i] = (__bf16)W3[k * 512 + n]; }
    if (i < 512 * 512) { int n = i / 512, k = i % 512; ws[W4T_OFF + i] = (__bf16)W4[k * 512 + n]; }
}

// ---------------------------------------------------------------------------
// Fused pipeline. Each wave owns 16 rows end-to-end -> zero cross-wave sync.
// ---------------------------------------------------------------------------
__global__ void __launch_bounds__(256)
fused_kernel(const float* __restrict__ x, const float* __restrict__ noise,
             const float* __restrict__ b1, const float* __restrict__ b2,
             const float* __restrict__ b3, const float* __restrict__ b4,
             const float* __restrict__ omega, const float* __restrict__ Kp,
             const float* __restrict__ alphap,
             const __bf16* __restrict__ W1t, const __bf16* __restrict__ W2t,
             const __bf16* __restrict__ W3t, const __bf16* __restrict__ W4t,
             float* __restrict__ out) {
    extern __shared__ __bf16 smem[];
    __bf16* tbuf = smem;                     // [TM][T_STRIDE] bf16
    __bf16* hbuf = smem + TM * T_STRIDE;     // [TM][H_STRIDE] bf16

    const int lane  = threadIdx.x & 31;
    const int wave  = threadIdx.x >> 5;
    const int half  = lane >> 4;          // 0: lanes 0-15, 1: lanes 16-31
    const int ln    = lane & 15;          // N index within a 16-wide tile
    const int khalf = half * 8;           // A-fragment K half select
    const int rowInTile = wave * 16 + ln; // row this lane's A-fragment feeds
    const int rowBase   = blockIdx.x * TM;

    const float Kc    = Kp[0];
    const float alpha = alphap[0];

    // ================= GEMM1: t = tanh(x @ W1 + b1)  [16 x 64]/wave =======
    v8f acc[4];
    acc[0] = v8f{}; acc[1] = v8f{}; acc[2] = v8f{}; acc[3] = v8f{};
    const float* xrow = x + (size_t)(rowBase + rowInTile) * E_DIM;
#pragma unroll 2
    for (int kc = 0; kc < E_DIM / 32; ++kc) {
        v8u A = a_frag_global(xrow + kc * 32, khalf);
        int kb = kc * 32 + half * 16;
#pragma unroll
        for (int nt = 0; nt < 4; ++nt)
            acc[nt] = wmma_bf16(A, b_frag(W1t, nt * 16 + ln, E_DIM, kb), acc[nt]);
    }
#pragma unroll
    for (int nt = 0; nt < 4; ++nt) {
        int n = nt * 16 + ln;
        float bias = b1[n];
#pragma unroll
        for (int v = 0; v < 8; ++v) {
            int m = v + half * 8;   // C layout: lane=n, vgpr v -> row m
            tbuf[(wave * 16 + m) * T_STRIDE + n] = (__bf16)fast_tanh(acc[nt][v] + bias);
        }
    }

    // ================= GEMM2: phases0 = t @ W2 + b2 ========================
    v8f ph[4];
    ph[0] = v8f{}; ph[1] = v8f{}; ph[2] = v8f{}; ph[3] = v8f{};
    const __bf16* trow = tbuf + rowInTile * T_STRIDE;
#pragma unroll
    for (int kc = 0; kc < 2; ++kc) {
        v8u A = a_frag_lds(trow + kc * 32, khalf);
        int kb = kc * 32 + half * 16;
#pragma unroll
        for (int nt = 0; nt < 4; ++nt)
            ph[nt] = wmma_bf16(A, b_frag(W2t, nt * 16 + ln, O_DIM, kb), ph[nt]);
    }
    float om[4];
#pragma unroll
    for (int nt = 0; nt < 4; ++nt) {
        int n = nt * 16 + ln;
        float bias = b2[n];
        om[nt] = omega[n];
#pragma unroll
        for (int v = 0; v < 8; ++v) ph[nt][v] += bias;
    }

    // ====== Kuramoto: 10 steps, register-resident, butterfly mean-field ====
    // Row m lives in one 16-lane half at one vgpr slot; its 64 oscillators
    // are the 4 n-tiles across those 16 lanes.
    for (int s = 0; s < N_STEPS; ++s) {
#pragma unroll
        for (int v = 0; v < 8; ++v) {
            float p = __sinf(ph[0][v]) + __sinf(ph[1][v]) +
                      __sinf(ph[2][v]) + __sinf(ph[3][v]);
            p += __shfl_xor(p, 1, 32);
            p += __shfl_xor(p, 2, 32);
            p += __shfl_xor(p, 4, 32);
            p += __shfl_xor(p, 8, 32);       // stays inside the 16-lane half
            float mean = p * (1.0f / 64.0f);
#pragma unroll
            for (int nt = 0; nt < 4; ++nt) {
                float c = __cosf(ph[nt][v]);
                ph[nt][v] += DT_STEP * (om[nt] + Kc * mean * c);
            }
        }
    }

    // phases -> LDS (bf16, A-fragment-friendly row-major), reuse tbuf
#pragma unroll
    for (int nt = 0; nt < 4; ++nt) {
        int n = nt * 16 + ln;
#pragma unroll
        for (int v = 0; v < 8; ++v) {
            int m = v + half * 8;
            tbuf[(wave * 16 + m) * T_STRIDE + n] = (__bf16)ph[nt][v];
        }
    }

    // ================= GEMM3: h = relu(phases @ W3 + b3)  [16 x 512]/wave ==
    const __bf16* prow = tbuf + rowInTile * T_STRIDE;
    v8u A30 = a_frag_lds(prow, khalf);
    v8u A31 = a_frag_lds(prow + 32, khalf);
#pragma unroll 1
    for (int nt = 0; nt < E_DIM / 16; ++nt) {
        int n = nt * 16 + ln;
        v8f g = v8f{};
        g = wmma_bf16(A30, b_frag(W3t, n, O_DIM, half * 16), g);
        g = wmma_bf16(A31, b_frag(W3t, n, O_DIM, 32 + half * 16), g);
        float bias = b3[n];
#pragma unroll
        for (int v = 0; v < 8; ++v) {
            float r = fmaxf(g[v] + bias, 0.0f);
            hbuf[(wave * 16 + v + half * 8) * H_STRIDE + n] = (__bf16)r;
        }
    }

    // ===== GEMM4 + epilogue: out = h @ W4 + b4, quaternion assembly ========
    const __bf16* hrow = hbuf + rowInTile * H_STRIDE;
#pragma unroll 1
    for (int nt = 0; nt < E_DIM / 16; ++nt) {
        int n = nt * 16 + ln;
        const __bf16* wcol = W4t + (size_t)n * E_DIM;
        v8f c = v8f{};
#pragma unroll 4
        for (int kc = 0; kc < E_DIM / 32; ++kc) {
            if (kc + 2 < E_DIM / 32)          // hide L2 latency on W4 column
                __builtin_prefetch(wcol + (kc + 2) * 32, 0, 1);
            v8u A = a_frag_lds(hrow + kc * 32, khalf);
            c = wmma_bf16(A, b_frag(W4t, n, E_DIM, kc * 32 + half * 16), c);
        }
        float bias = b4[n];
        float fm = __sinf(alpha * (float)n);  // fractal modulation sin(alpha*e)
#pragma unroll
        for (int v = 0; v < 8; ++v) {
            int R = rowBase + wave * 16 + v + half * 8;
            int idx = R * E_DIM + n;
            float im = noise[idx] * fm;
            *(float4*)(out + (size_t)idx * 4) = make_float4(c[v] + bias, im, im, im);
        }
    }
}

// ---------------------------------------------------------------------------
extern "C" void kernel_launch(void* const* d_in, const int* in_sizes, int n_in,
                              void* d_out, int out_size, void* d_ws, size_t ws_size,
                              hipStream_t stream) {
    const float* x      = (const float*)d_in[0];
    const float* noise  = (const float*)d_in[1];
    const float* W1     = (const float*)d_in[2];
    const float* b1     = (const float*)d_in[3];
    const float* W2     = (const float*)d_in[4];
    const float* b2     = (const float*)d_in[5];
    const float* W3     = (const float*)d_in[6];
    const float* b3     = (const float*)d_in[7];
    const float* W4     = (const float*)d_in[8];
    const float* b4     = (const float*)d_in[9];
    const float* omega  = (const float*)d_in[10];
    const float* Kp     = (const float*)d_in[11];
    const float* alphap = (const float*)d_in[12];

    __bf16* ws = (__bf16*)d_ws;

    // bf16 transposed weights into workspace (648 KB, stays in 192 MB L2)
    convert_weights<<<(512 * 512 + 255) / 256, 256, 0, stream>>>(W1, W2, W3, W4, ws);

    const __bf16* W1t = ws + W1T_OFF;
    const __bf16* W2t = ws + W2T_OFF;
    const __bf16* W3t = ws + W3T_OFF;
    const __bf16* W4t = ws + W4T_OFF;

    const int rows = 8 * 4096;
    const size_t lds_bytes = (size_t)(TM * T_STRIDE + TM * H_STRIDE) * sizeof(__bf16);
    fused_kernel<<<rows / TM, 256, lds_bytes, stream>>>(
        x, noise, b1, b2, b3, b4, omega, Kp, alphap,
        W1t, W2t, W3t, W4t, (float*)d_out);
}
